// PacketDirectionPatternModule_22892175688161
// MI455X (gfx1250) — compile-verified, matrix operand order checked by reading, not run on hardware
//
#include <hip/hip_runtime.h>
#include <hip/hip_bf16.h>
#include <math.h>

// ---------------------------------------------------------------------------
// CDNA5 / gfx1250: wave32, WMMA 16x16x32 bf16 (f32 accumulate)
// ---------------------------------------------------------------------------
typedef __attribute__((ext_vector_type(16))) __bf16 bf16x16;
typedef __attribute__((ext_vector_type(8)))  float  f32x8;

#define WMMA_BF16(a, b, c) \
  __builtin_amdgcn_wmma_f32_16x16x32_bf16(false, (a), false, (b), (short)0, (c), false, false)

static constexpr int Bb = 16;
static constexpr int Tt = 1024;
static constexpr int Dd = 128;
static constexpr int HD = 32;       // head dim
static constexpr int QT = Tt / 16;  // query tiles per (b,h)

// Load one 16x16x32 bf16 A/B fragment for a lane whose row/column is
// contiguous along K in memory (row-major, K-contiguous).
// Layout (ISA 7.12.2, 16-bit A 16x32): lanes 0-15 hold K = {0..7,16..23},
// lanes 16-31 hold K = {8..15,24..31}; element e -> K = (e<8? e : e+8)+half*8.
// Both 8-element runs are 16B contiguous -> two b128 loads per lane.
__device__ __forceinline__ bf16x16 load_frag(const __bf16* rowptr, int khalf) {
  const __bf16* p = rowptr + khalf * 8;
  bf16x16 f;
#pragma unroll
  for (int e = 0; e < 8; ++e) f[e] = p[e];
#pragma unroll
  for (int e = 0; e < 8; ++e) f[8 + e] = p[16 + e];
  return f;
}

// ---------------------------------------------------------------------------
// DPP-based cross-lane reductions (no LDS, pure VALU, dual-issuable).
// 16-lane all-reduce: xor1 (quad_perm 0xB1), xor2 (quad_perm 0x4E),
// row_half_mirror (0x141, == xor7 once quads are uniform),
// row_mirror (0x140, == xor15 once 8-groups are uniform).
// ---------------------------------------------------------------------------
template <int CTRL>
__device__ __forceinline__ float dpp_mov(float v) {
  return __int_as_float(__builtin_amdgcn_update_dpp(
      0, __float_as_int(v), CTRL, 0xF, 0xF, true));
}

__device__ __forceinline__ float red16_max(float v) {
  v = fmaxf(v, dpp_mov<0x0B1>(v));
  v = fmaxf(v, dpp_mov<0x04E>(v));
  v = fmaxf(v, dpp_mov<0x141>(v));
  v = fmaxf(v, dpp_mov<0x140>(v));
  return v;
}
__device__ __forceinline__ float red16_sum(float v) {
  v += dpp_mov<0x0B1>(v);
  v += dpp_mov<0x04E>(v);
  v += dpp_mov<0x141>(v);
  v += dpp_mov<0x140>(v);
  return v;
}
// Full wave32 sum: 16-lane DPP reduce + one v_permlanex16 cross-half step
// (halves are uniform by then, so any lane select is valid).
__device__ __forceinline__ float red32_sum(float v) {
  v = red16_sum(v);
  int vi = __float_as_int(v);
  int sw = __builtin_amdgcn_permlanex16(vi, vi, 0, 0, false, false);
  return v + __int_as_float(sw);
}

// ---------------------------------------------------------------------------
// fp32 -> bf16 weight conversion
// ---------------------------------------------------------------------------
__global__ void cvt_bf16(const float* __restrict__ s, __bf16* __restrict__ d, int n) {
  int i = blockIdx.x * blockDim.x + threadIdx.x;
  for (; i < n; i += gridDim.x * blockDim.x) d[i] = (__bf16)s[i];
}

// ---------------------------------------------------------------------------
// Extract: RLE features + embedding gathers + dc projection.
// One block per batch row; serial scan is negligible vs the attention work.
// ---------------------------------------------------------------------------
__global__ __launch_bounds__(256)
void extract_kernel(const float* __restrict__ x,
                    const float* __restrict__ trans_emb,
                    const float* __restrict__ blen_emb,
                    const float* __restrict__ bpos_emb,
                    const float* __restrict__ dc_w,
                    const float* __restrict__ dc_b,
                    __bf16* __restrict__ tf, __bf16* __restrict__ bff,
                    float* __restrict__ dcp) {
  __shared__ int   s_dir[Tt];
  __shared__ short s_start[Tt];
  __shared__ short s_len[Tt];
  __shared__ float s_inc;
  const int b = blockIdx.x, tid = threadIdx.x;

  for (int t = tid; t < Tt; t += 256)
    s_dir[t] = (x[(size_t)b * Tt + t] > 0.f) ? 1 : 0;
  __syncthreads();

  if (tid == 0) {
    int cnt = 0, cur = 0;
    for (int t = 0; t < Tt; ++t) {
      if (t > 0 && s_dir[t] != s_dir[t - 1]) cur = t;
      s_start[t] = (short)cur;
      cnt += s_dir[t];
    }
    int nxt = Tt;
    for (int t = Tt - 1; t >= 0; --t) {
      s_len[t] = (short)(nxt - s_start[t]);
      if (t > 0 && s_dir[t] != s_dir[t - 1]) nxt = t;
    }
    s_inc = (float)cnt / (float)Tt;
  }
  __syncthreads();

  const float inc = s_inc, ov = 1.f - inc;
  if (tid < Dd) {
    dcp[b * Dd + tid] = inc * (dc_w[tid * 4 + 0] + dc_w[tid * 4 + 2]) +
                        ov  * (dc_w[tid * 4 + 1] + dc_w[tid * 4 + 3]) + dc_b[tid];
  }
  for (int i = tid; i < Tt * Dd; i += 256) {
    const int t = i >> 7, d = i & 127;
    const int trid = (t == 0) ? 0 : (s_dir[t - 1] * 2 + s_dir[t]);
    tf[((size_t)b * Tt + t) * Dd + d] = (__bf16)trans_emb[trid * Dd + d];
    int L = (int)s_len[t]; if (L > 31) L = 31;
    int P = ((int)s_start[t] * 64) / Tt; if (P > 63) P = 63;
    bff[((size_t)b * Tt + t) * Dd + d] =
        (__bf16)(0.5f * (blen_emb[L * Dd + d] + bpos_emb[P * Dd + d]));
  }
}

// ---------------------------------------------------------------------------
// Generic bf16 GEMM with bias:  C(M,N) = A(M,K) @ W(N,K)^T + bias
// One 32x16 output tile per wave (two WMMAs share one B fragment),
// 8 waves/block, K stepped by 32.
// ---------------------------------------------------------------------------
template <bool OUT_BF16>
__global__ __launch_bounds__(256)
void gemm_bias(const __bf16* __restrict__ A, int lda,
               const __bf16* __restrict__ W, int ldw,
               const float* __restrict__ bias,
               void* __restrict__ Cout, int M, int N, int K) {
  const int wave = threadIdx.x >> 5;
  const int lane = threadIdx.x & 31;
  const int l16 = lane & 15, half = lane >> 4;
  const int nt_cnt = N >> 4;
  const int tile = blockIdx.x * 8 + wave;          // wave-uniform
  if (tile >= (M >> 5) * nt_cnt) return;           // EXEC stays all-ones
  const int mt = tile / nt_cnt, nt = tile % nt_cnt;

  const __bf16* arow0 = A + (size_t)(mt * 32 + l16) * lda;
  const __bf16* arow1 = arow0 + (size_t)16 * lda;
  const __bf16* wrow  = W + (size_t)(nt * 16 + l16) * ldw;

  f32x8 acc0 = {}, acc1 = {};
  for (int k0 = 0; k0 < K; k0 += 32) {
    __builtin_prefetch(arow0 + k0 + 64, 0, 0);     // global_prefetch_b8
    bf16x16 bfrag = load_frag(wrow + k0, half);
    bf16x16 a0 = load_frag(arow0 + k0, half);
    bf16x16 a1 = load_frag(arow1 + k0, half);
    acc0 = WMMA_BF16(a0, bfrag, acc0);
    acc1 = WMMA_BF16(a1, bfrag, acc1);
  }

  const float bv = bias[nt * 16 + l16];
  const int n = nt * 16 + l16;
#pragma unroll
  for (int r = 0; r < 8; ++r) {
    const int m0 = mt * 32 + r + 8 * half;         // C/D VGPR layout
    const int m1 = m0 + 16;
    const float v0 = acc0[r] + bv, v1 = acc1[r] + bv;
    if (OUT_BF16) {
      ((__bf16*)Cout)[(size_t)m0 * N + n] = (__bf16)v0;
      ((__bf16*)Cout)[(size_t)m1 * N + n] = (__bf16)v1;
    } else {
      ((float*)Cout)[(size_t)m0 * N + n] = v0;
      ((float*)Cout)[(size_t)m1 * N + n] = v1;
    }
  }
}

// ---------------------------------------------------------------------------
// Fused flash-attention: one wave per (b, h, 16-query tile), 8 waves/block.
// hd=32 = one WMMA K-step for Q*K^T. Online softmax via DPP reductions;
// P re-laid out via a private 1KB LDS slice per wave (same-wave LDS ops are
// architecturally in-order, so only a compiler fence is needed, no barrier).
// Scores never touch HBM.
// qkv layout: (B*T, 384) = [Q | K | V], head h at column h*32 within each.
// ---------------------------------------------------------------------------
__global__ __launch_bounds__(256)
void attention_kernel(const __bf16* __restrict__ qkv,
                      __bf16* __restrict__ outp, float scale) {
  __shared__ __bf16 Pld[8][16 * 32];
  const int wave = threadIdx.x >> 5;
  const int lane = threadIdx.x & 31;
  const int l16 = lane & 15, half = lane >> 4;
  const int tileId = blockIdx.x * 8 + wave;
  const int qt = tileId & (QT - 1);
  const int h  = (tileId >> 6) & 3;
  const int b  = tileId >> 8;
  __bf16* P = Pld[wave];

  const __bf16* base = qkv + (size_t)b * Tt * 384;
  const __bf16* Qp = base + h * HD;
  const __bf16* Kp = base + Dd + h * HD;
  const __bf16* Vp = base + 2 * Dd + h * HD;

  const bf16x16 qf = load_frag(Qp + (size_t)(qt * 16 + l16) * 384, half);

  float mi[8], li[8];
#pragma unroll
  for (int r = 0; r < 8; ++r) { mi[r] = -1e30f; li[r] = 0.f; }
  f32x8 o0 = {}, o1 = {};

  for (int kc = 0; kc < Tt; kc += 32) {
    bf16x16 k0f = load_frag(Kp + (size_t)(kc + l16) * 384, half);
    bf16x16 k1f = load_frag(Kp + (size_t)(kc + 16 + l16) * 384, half);
    f32x8 z = {};
    f32x8 s0 = WMMA_BF16(qf, k0f, z);
    f32x8 s1 = WMMA_BF16(qf, k1f, z);

#pragma unroll
    for (int r = 0; r < 8; ++r) {
      const float x0 = s0[r] * scale, x1 = s1[r] * scale;
      const float mnew = fmaxf(mi[r], red16_max(fmaxf(x0, x1)));
      const float p0 = __expf(x0 - mnew), p1 = __expf(x1 - mnew);
      const float rs = red16_sum(p0 + p1);
      const float alpha = __expf(mi[r] - mnew);
      li[r] = li[r] * alpha + rs;
      mi[r] = mnew;
      o0[r] *= alpha;
      o1[r] *= alpha;
      P[(r + 8 * half) * 32 + l16]      = (__bf16)p0;
      P[(r + 8 * half) * 32 + 16 + l16] = (__bf16)p1;
    }
    asm volatile("" ::: "memory");   // compiler fence: HW keeps LDS in order

    const bf16x16 pf = load_frag(P + l16 * 32, half);     // P as A-fragment
    bf16x16 v0, v1;                                        // V as B-fragments
#pragma unroll
    for (int e = 0; e < 16; ++e) {
      const int kk = ((e < 8) ? e : e + 8) + half * 8;
      const __bf16* vr = Vp + (size_t)(kc + kk) * 384;
      v0[e] = vr[l16];
      v1[e] = vr[16 + l16];
    }
    o0 = WMMA_BF16(pf, v0, o0);
    o1 = WMMA_BF16(pf, v1, o1);
    asm volatile("" ::: "memory");   // keep next-iter P stores after pf load
  }

#pragma unroll
  for (int r = 0; r < 8; ++r) {
    const int tr = qt * 16 + r + 8 * half;
    const float invl = 1.f / li[r];
    __bf16* orow = outp + ((size_t)b * Tt + tr) * Dd + h * HD;
    orow[l16]      = (__bf16)(o0[r] * invl);
    orow[16 + l16] = (__bf16)(o1[r] * invl);
  }
}

// ---------------------------------------------------------------------------
// comb = [ta | ba | broadcast(dcp)]  (B*T, 384) bf16
// ---------------------------------------------------------------------------
__global__ void comb_kernel(const __bf16* __restrict__ ta,
                            const __bf16* __restrict__ ba,
                            const float* __restrict__ dcp,
                            __bf16* __restrict__ comb) {
  size_t i = (size_t)blockIdx.x * blockDim.x + threadIdx.x;
  const size_t total = (size_t)Bb * Tt * 384;
  for (; i < total; i += (size_t)gridDim.x * blockDim.x) {
    const size_t bt = i / 384;
    const int c = (int)(i % 384);
    __bf16 v;
    if (c < 128)      v = ta[bt * 128 + c];
    else if (c < 256) v = ba[bt * 128 + (c - 128)];
    else {
      const int b = (int)(bt >> 10);   // T = 1024
      v = (__bf16)dcp[b * 128 + (c - 256)];
    }
    comb[i] = v;
  }
}

// ---------------------------------------------------------------------------
// Row LayerNorm + exact GELU. One wave per row (NC = 256 or 128).
// ---------------------------------------------------------------------------
template <int NC, bool OUT_BF16>
__global__ __launch_bounds__(256)
void ln_gelu(const float* __restrict__ in, const float* __restrict__ gw,
             const float* __restrict__ bw, void* __restrict__ outp, int rows) {
  const int wave = threadIdx.x >> 5, lane = threadIdx.x & 31;
  const int row = blockIdx.x * 8 + wave;
  if (row >= rows) return;
  const float* rp = in + (size_t)row * NC;
  constexpr int PER = NC / 32;
  float v[PER];
  float s = 0.f;
#pragma unroll
  for (int i = 0; i < PER; ++i) { v[i] = rp[lane + i * 32]; s += v[i]; }
  s = red32_sum(s);
  const float mean = s * (1.f / NC);
  float vs = 0.f;
#pragma unroll
  for (int i = 0; i < PER; ++i) { const float d = v[i] - mean; vs += d * d; }
  vs = red32_sum(vs);
  const float rstd = rsqrtf(vs * (1.f / NC) + 1e-5f);
#pragma unroll
  for (int i = 0; i < PER; ++i) {
    const int c = lane + i * 32;
    const float y = (v[i] - mean) * rstd * gw[c] + bw[c];
    const float g = 0.5f * y * (1.f + erff(y * 0.70710678118654752f));
    if (OUT_BF16) ((__bf16*)outp)[(size_t)row * NC + c] = (__bf16)g;
    else          ((float*) outp)[(size_t)row * NC + c] = g;
  }
}

// ---------------------------------------------------------------------------
// Host-side orchestration
// ---------------------------------------------------------------------------
extern "C" void kernel_launch(void* const* d_in, const int* in_sizes, int n_in,
                              void* d_out, int out_size, void* d_ws, size_t ws_size,
                              hipStream_t stream) {
  const float* x         = (const float*)d_in[0];
  const float* trans_emb = (const float*)d_in[1];
  const float* blen_emb  = (const float*)d_in[2];
  const float* bpos_emb  = (const float*)d_in[3];
  const float* dc_w      = (const float*)d_in[4];
  const float* dc_b      = (const float*)d_in[5];
  const float* ta_in_w   = (const float*)d_in[6];
  const float* ta_in_b   = (const float*)d_in[7];
  const float* ta_out_w  = (const float*)d_in[8];
  const float* ta_out_b  = (const float*)d_in[9];
  const float* ba_in_w   = (const float*)d_in[10];
  const float* ba_in_b   = (const float*)d_in[11];
  const float* ba_out_w  = (const float*)d_in[12];
  const float* ba_out_b  = (const float*)d_in[13];
  const float* f1_w      = (const float*)d_in[14];
  const float* f1_b      = (const float*)d_in[15];
  const float* ln1_g     = (const float*)d_in[16];
  const float* ln1_b     = (const float*)d_in[17];
  const float* f2_w      = (const float*)d_in[18];
  const float* f2_b      = (const float*)d_in[19];
  const float* ln2_g     = (const float*)d_in[20];
  const float* ln2_b     = (const float*)d_in[21];

  // --- workspace arena ---
  char* wsp = (char*)d_ws;
  size_t off = 0;
  auto alloc = [&](size_t bytes) -> void* {
    void* p = wsp + off;
    off += (bytes + 255) & ~(size_t)255;
    return p;
  };
  const size_t BT = (size_t)Bb * Tt;                    // 16384
  __bf16* w_ta_in  = (__bf16*)alloc(384 * 128 * 2);
  __bf16* w_ta_out = (__bf16*)alloc(128 * 128 * 2);
  __bf16* w_ba_in  = (__bf16*)alloc(384 * 128 * 2);
  __bf16* w_ba_out = (__bf16*)alloc(128 * 128 * 2);
  __bf16* w_f1     = (__bf16*)alloc(256 * 384 * 2);
  __bf16* w_f2     = (__bf16*)alloc(128 * 256 * 2);
  float*  dcp      = (float*) alloc(Bb * Dd * 4);
  __bf16* tf       = (__bf16*)alloc(BT * Dd * 2);
  __bf16* bff      = (__bf16*)alloc(BT * Dd * 2);
  __bf16* qkv      = (__bf16*)alloc(BT * 384 * 2);      // reused: qkv_t, qkv_b, comb
  __bf16* atto     = (__bf16*)alloc(BT * Dd * 2);       // reused by both attentions
  __bf16* ta       = (__bf16*)alloc(BT * Dd * 2);
  __bf16* ba       = (__bf16*)alloc(BT * Dd * 2);
  float*  f1o      = (float*) alloc(BT * 256 * 4);      // reused for f2 output
  __bf16* hbuf     = (__bf16*)alloc(BT * 256 * 2);
  (void)ws_size; (void)n_in; (void)in_sizes; (void)out_size;

  // --- weights to bf16 ---
  cvt_bf16<<<192, 256, 0, stream>>>(ta_in_w,  w_ta_in,  384 * 128);
  cvt_bf16<<<64,  256, 0, stream>>>(ta_out_w, w_ta_out, 128 * 128);
  cvt_bf16<<<192, 256, 0, stream>>>(ba_in_w,  w_ba_in,  384 * 128);
  cvt_bf16<<<64,  256, 0, stream>>>(ba_out_w, w_ba_out, 128 * 128);
  cvt_bf16<<<384, 256, 0, stream>>>(f1_w,     w_f1,     256 * 384);
  cvt_bf16<<<128, 256, 0, stream>>>(f2_w,     w_f2,     128 * 256);

  // --- extract features ---
  extract_kernel<<<Bb, 256, 0, stream>>>(x, trans_emb, blen_emb, bpos_emb,
                                         dc_w, dc_b, tf, bff, dcp);

  const float scale = 0.17677669529663687f;   // 1/sqrt(32)

  // --- transition attention ---
  gemm_bias<true><<<1536, 256, 0, stream>>>(tf, 128, w_ta_in, 128, ta_in_b,
                                            qkv, (int)BT, 384, 128);
  attention_kernel<<<512, 256, 0, stream>>>(qkv, atto, scale);
  gemm_bias<true><<<512, 256, 0, stream>>>(atto, 128, w_ta_out, 128, ta_out_b,
                                           ta, (int)BT, 128, 128);

  // --- burst attention (reuses qkv and atto buffers) ---
  gemm_bias<true><<<1536, 256, 0, stream>>>(bff, 128, w_ba_in, 128, ba_in_b,
                                            qkv, (int)BT, 384, 128);
  attention_kernel<<<512, 256, 0, stream>>>(qkv, atto, scale);
  gemm_bias<true><<<512, 256, 0, stream>>>(atto, 128, w_ba_out, 128, ba_out_b,
                                           ba, (int)BT, 128, 128);

  // --- combine + MLP head ---
  comb_kernel<<<2048, 256, 0, stream>>>(ta, ba, dcp, qkv);   // comb into qkv region
  gemm_bias<false><<<1024, 256, 0, stream>>>(qkv, 384, w_f1, 384, f1_b,
                                             f1o, (int)BT, 256, 384);
  ln_gelu<256, true><<<2048, 256, 0, stream>>>(f1o, ln1_g, ln1_b, hbuf, (int)BT);
  gemm_bias<false><<<512, 256, 0, stream>>>(hbuf, 256, w_f2, 256, f2_b,
                                            f1o /*reuse*/, (int)BT, 128, 256);
  ln_gelu<128, false><<<2048, 256, 0, stream>>>(f1o, ln2_g, ln2_b, d_out, (int)BT);
}
